// DecoderSRNN_81149112091056
// MI455X (gfx1250) — matrix operand order repaired and checked by reading, not compiled, not run on hardware
//
#include <hip/hip_runtime.h>

#define BATCHN 64
#define HIDDENN 512
#define VOCABN 32000
#define SEQN 64
#define NSTACKN 2
#define NACTN 3
#define SSIZEN 64
#define ESIZEN 128
#define KMID 2048  // emb(512) + hid(512) + 2 stacks * DEPTH*ESIZE(512)

typedef __attribute__((ext_vector_type(16))) __bf16 v16bf;
typedef __attribute__((ext_vector_type(8)))  __bf16 v8bf;
typedef __attribute__((ext_vector_type(8)))  float  v8f;

__device__ __forceinline__ __bf16 f2bf(float f) {
  union { float f; unsigned u; } in; in.f = f;
  unsigned r = in.u + 0x7FFFu + ((in.u >> 16) & 1u);   // round-to-nearest-even
  union { unsigned short s; __bf16 b; } out; out.s = (unsigned short)(r >> 16);
  return out.b;
}

__device__ __forceinline__ float sigmoidf_(float x) { return 1.0f / (1.0f + __expf(-x)); }

// A-matrix (16x32 MxK bf16) fragment: lane half 0 -> K = {k0..k0+7, k0+16..k0+23},
// lane half 1 -> K = {k0+8..k0+15, k0+24..k0+31}  (ISA 7.12.2, interleaved)
__device__ __forceinline__ v16bf load_frag_a(const __bf16* __restrict__ p, int hf) {
  union { v8bf h[2]; v16bf v; } u;
  u.h[0] = *reinterpret_cast<const v8bf*>(p + hf * 8);
  u.h[1] = *reinterpret_cast<const v8bf*>(p + 16 + hf * 8);
  return u.v;
}

// B-matrix (32x16 KxN bf16) fragment: lane half 0 -> K = {k0..k0+7, k0+16..k0+23},
// half 1 -> K = {k0+8..k0+15, k0+24..k0+31}, interleaved (same as A)
__device__ __forceinline__ v16bf load_frag_b(const __bf16* __restrict__ p, int hf) {
  union { v8bf h[2]; v16bf v; } u;
  u.h[0] = *reinterpret_cast<const v8bf*>(p + hf * 8);
  u.h[1] = *reinterpret_cast<const v8bf*>(p + 16 + hf * 8);
  return u.v;
}

// ---------------- one-time prep: bf16 weights (N-major), combined bias, stack init
__global__ void prep_kernel(const float* __restrict__ W_out, const float* __restrict__ W_i2h,
                            const float* __restrict__ W_h2h, const float* __restrict__ W_s2h,
                            const float* __restrict__ b_i2h, const float* __restrict__ b_h2h,
                            const float* __restrict__ b_s2h, const float* __restrict__ empty_elem,
                            __bf16* __restrict__ WOUT, __bf16* __restrict__ BCAT,
                            float* __restrict__ biasc, float* __restrict__ st0)
{
  long stride = (long)gridDim.x * blockDim.x;
  long gid = (long)blockIdx.x * blockDim.x + threadIdx.x;
  for (long i = gid; i < (long)VOCABN * HIDDENN; i += stride) WOUT[i] = f2bf(W_out[i]);
  // BCAT[n][k], n-major, K = [W_i2h row | W_h2h row | W_s2h[0] row | W_s2h[1] row]
  for (long i = gid; i < (long)HIDDENN * KMID; i += stride) {
    int n = (int)(i >> 11), k = (int)(i & (KMID - 1));
    float v;
    if (k < 512)        v = W_i2h[n * 512 + k];
    else if (k < 1024)  v = W_h2h[n * 512 + (k - 512)];
    else if (k < 1536)  v = W_s2h[n * 512 + (k - 1024)];
    else                v = W_s2h[512 * 512 + n * 512 + (k - 1536)];
    BCAT[i] = f2bf(v);
  }
  for (long i = gid; i < HIDDENN; i += stride)
    biasc[i] = b_i2h[i] + b_h2h[i] + b_s2h[i] + b_s2h[HIDDENN + i];
  for (long i = gid; i < (long)BATCHN * NSTACKN * SSIZEN * ESIZEN; i += stride)
    st0[i] = empty_elem[i & (ESIZEN - 1)];
}

// ---------------- gates: acts[b][n][a], push_val[b][n][e] from OLD hidden (VALU, tiny)
__global__ void gates_kernel(const float* __restrict__ hid, const float* __restrict__ W_h2s,
                             const float* __restrict__ b_h2s, const float* __restrict__ W_h2a,
                             const float* __restrict__ b_h2a,
                             float* __restrict__ push_val, float* __restrict__ acts)
{
  int b = blockIdx.x, tid = threadIdx.x;
  const float* h = hid + b * HIDDENN;
  int n = tid >> 7, e = tid & 127;
  const float* w = W_h2s + (n * ESIZEN + e) * HIDDENN;
  float d = 0.f;
  for (int k = 0; k < HIDDENN; ++k) d += h[k] * w[k];
  push_val[b * (NSTACKN * ESIZEN) + tid] = sigmoidf_(d + b_h2s[n * ESIZEN + e]);
  if (tid < NSTACKN * NACTN) {
    int nn = tid / NACTN, a = tid % NACTN;
    const float* wa = W_h2a + (nn * NACTN + a) * HIDDENN;
    float da = 0.f;
    for (int k = 0; k < HIDDENN; ++k) da += h[k] * wa[k];
    acts[b * (NSTACKN * NACTN) + tid] = da + b_h2a[nn * NACTN + a];
  }
}

// ---------------- build bf16 A = [emb | hid | stack0_top4 | stack1_top4], 64 x 2048
__global__ void aprep_kernel(const int* __restrict__ tok, const float* __restrict__ embedding,
                             const float* __restrict__ hid, const float* __restrict__ st,
                             __bf16* __restrict__ A)
{
  int gid = blockIdx.x * blockDim.x + threadIdx.x;  // 64*2048 threads
  int b = gid >> 11, k = gid & 2047;
  float v;
  if (k < 512)       v = embedding[(long)tok[b] * HIDDENN + k];
  else if (k < 1024) v = hid[b * HIDDENN + (k - 512)];
  else {
    int idx = k - 1024;
    int n = idx >> 9, r = idx & 511;  // top DEPTH*ESIZE elems of stack n are contiguous
    v = st[((b * NSTACKN + n) * SSIZEN) * ESIZEN + r];
  }
  A[gid] = f2bf(v);
}

// ---------------- stack shift / gate blend, elementwise over [64,2,64,128]
__global__ void stack_kernel(const float* __restrict__ stin, const float* __restrict__ acts,
                             const float* __restrict__ push_val, const float* __restrict__ empty_elem,
                             float* __restrict__ stout)
{
  int gid = blockIdx.x * blockDim.x + threadIdx.x;  // 1,048,576 threads
  int e = gid & 127, s = (gid >> 7) & 63, n = (gid >> 13) & 1, b = gid >> 14;
  int abase = b * (NSTACKN * NACTN) + n * NACTN;
  float ppush = acts[abase + 0], ppop = acts[abase + 1], pnoop = acts[abase + 2];
  float out;
  if (s == 0)                out = ppush * push_val[b * (NSTACKN * ESIZEN) + n * ESIZEN + e];
  else if (s == SSIZEN - 1)  out = empty_elem[e];
  else                       out = ppush * stin[gid - ESIZEN] + ppop * stin[gid + ESIZEN] + pnoop * stin[gid];
  stout[gid] = out;
}

// ---------------- mid GEMM: [64,2048]bf16 x [2048,512] -> sigmoid -> hidden (f32 + bf16)
__global__ void midgemm_kernel(const __bf16* __restrict__ A, const __bf16* __restrict__ Bt,
                               const float* __restrict__ biasc,
                               float* __restrict__ hidO, __bf16* __restrict__ hidObf)
{
  int gw = (blockIdx.x * blockDim.x + threadIdx.x) >> 5;  // 32 waves total
  int lane = threadIdx.x & 31;
  int lm = lane & 15, hf = lane >> 4;
  int m0 = (gw & 3) * 16;
  int n0 = (gw >> 2) * 64;
  v8f acc[4] = {v8f{}, v8f{}, v8f{}, v8f{}};
  const __bf16* arow = A + (m0 + lm) * KMID;
  const __bf16* brow[4];
#pragma unroll
  for (int i = 0; i < 4; ++i) brow[i] = Bt + (long)(n0 + i * 16 + lm) * KMID;
  for (int k0 = 0; k0 < KMID; k0 += 32) {
    v16bf af = load_frag_a(arow + k0, hf);
#pragma unroll
    for (int i = 0; i < 4; ++i) {
      v16bf bfb = load_frag_b(brow[i] + k0, hf);
      acc[i] = __builtin_amdgcn_wmma_f32_16x16x32_bf16(false, af, false, bfb,
                                                       (short)0, acc[i], false, false);
    }
  }
#pragma unroll
  for (int i = 0; i < 4; ++i) {
    int nn = n0 + i * 16 + lm;
    float bias = biasc[nn];
#pragma unroll
    for (int j = 0; j < 8; ++j) {
      int m = m0 + j + hf * 8;
      float sv = sigmoidf_(acc[i][j] + bias);
      hidO[m * HIDDENN + nn] = sv;
      hidObf[m * HIDDENN + nn] = f2bf(sv);
    }
  }
}

// ---------------- vocab GEMM: [64,512]bf16 x [512,32000] + b_out -> raw logits
// One wave owns ALL of M (4 tiles) x 64 columns (4 tiles): 16 WMMA per k-step vs
// 16 b128 loads (1:1), and W_out bf16 (32.8MB, L2-resident) is read exactly once/step.
__global__ void logits_kernel(const __bf16* __restrict__ A, const __bf16* __restrict__ W,
                              const float* __restrict__ b_out, float* __restrict__ out)
{
  int gw = (blockIdx.x * blockDim.x + threadIdx.x) >> 5;  // 500 waves total
  int lane = threadIdx.x & 31;
  int lm = lane & 15, hf = lane >> 4;
  int n0 = gw * 64;
  v8f acc[4][4];
#pragma unroll
  for (int mi = 0; mi < 4; ++mi)
#pragma unroll
    for (int ni = 0; ni < 4; ++ni) acc[mi][ni] = v8f{};
  const __bf16* arow[4];
  const __bf16* brow[4];
#pragma unroll
  for (int mi = 0; mi < 4; ++mi) arow[mi] = A + (mi * 16 + lm) * HIDDENN;
#pragma unroll
  for (int ni = 0; ni < 4; ++ni) brow[ni] = W + (long)(n0 + ni * 16 + lm) * HIDDENN;
  for (int k0 = 0; k0 < HIDDENN; k0 += 32) {
    v16bf af[4], bfv[4];
#pragma unroll
    for (int mi = 0; mi < 4; ++mi) af[mi] = load_frag_a(arow[mi] + k0, hf);
#pragma unroll
    for (int ni = 0; ni < 4; ++ni) bfv[ni] = load_frag_b(brow[ni] + k0, hf);
#pragma unroll
    for (int ni = 0; ni < 4; ++ni)
#pragma unroll
      for (int mi = 0; mi < 4; ++mi)
        acc[mi][ni] = __builtin_amdgcn_wmma_f32_16x16x32_bf16(false, af[mi], false, bfv[ni],
                                                              (short)0, acc[mi][ni], false, false);
  }
#pragma unroll
  for (int ni = 0; ni < 4; ++ni) {
    int v = n0 + ni * 16 + lm;
    float bias = b_out[v];
#pragma unroll
    for (int mi = 0; mi < 4; ++mi)
#pragma unroll
      for (int j = 0; j < 8; ++j) {
        int m = mi * 16 + j + hf * 8;
        out[(long)m * VOCABN + v] = acc[mi][ni][j] + bias;
      }
  }
}

// ---------------- per-row max + logsumexp over 32000
__global__ void rowstats_kernel(const float* __restrict__ out, float* __restrict__ lse)
{
  __shared__ float red[256];
  int b = blockIdx.x, tid = threadIdx.x;
  const float* row = out + (long)b * VOCABN;
  float m = -3.4e38f;
  for (int i = tid; i < VOCABN; i += 256) m = fmaxf(m, row[i]);
  red[tid] = m; __syncthreads();
  for (int s = 128; s > 0; s >>= 1) {
    if (tid < s) red[tid] = fmaxf(red[tid], red[tid + s]);
    __syncthreads();
  }
  m = red[0]; __syncthreads();
  float sum = 0.f;
  for (int i = tid; i < VOCABN; i += 256) sum += __expf(row[i] - m);
  red[tid] = sum; __syncthreads();
  for (int s = 128; s > 0; s >>= 1) {
    if (tid < s) red[tid] += red[tid + s];
    __syncthreads();
  }
  if (tid == 0) lse[b] = m + __logf(red[0]);
}

// ---------------- out = logits - lse[row]
__global__ void logsub_kernel(float* __restrict__ out, const float* __restrict__ lse)
{
  long stride = (long)gridDim.x * blockDim.x;
  for (long i = (long)blockIdx.x * blockDim.x + threadIdx.x;
       i < (long)BATCHN * VOCABN; i += stride)
    out[i] -= lse[i / VOCABN];
}

__global__ void copyhid_kernel(const float* __restrict__ hid, float* __restrict__ dst)
{
  int gid = blockIdx.x * blockDim.x + threadIdx.x;  // 32768
  dst[gid] = hid[gid];
}

extern "C" void kernel_launch(void* const* d_in, const int* in_sizes, int n_in,
                              void* d_out, int out_size, void* d_ws, size_t ws_size,
                              hipStream_t stream) {
  (void)in_sizes; (void)n_in; (void)out_size; (void)ws_size;
  const int*   inputs    = (const int*)d_in[0];
  const float* hidden0   = (const float*)d_in[1];
  const float* embedding = (const float*)d_in[3];
  const float* W_i2h = (const float*)d_in[4],  *b_i2h = (const float*)d_in[5];
  const float* W_h2h = (const float*)d_in[6],  *b_h2h = (const float*)d_in[7];
  const float* W_h2a = (const float*)d_in[8],  *b_h2a = (const float*)d_in[9];
  const float* W_h2s = (const float*)d_in[10], *b_h2s = (const float*)d_in[11];
  const float* W_s2h = (const float*)d_in[12], *b_s2h = (const float*)d_in[13];
  const float* W_out = (const float*)d_in[14], *b_out = (const float*)d_in[15];
  const float* empty_elem = (const float*)d_in[16];

  char* ws = (char*)d_ws;
  size_t off = 0;
  auto alloc = [&](size_t bytes) { size_t o = off; off += (bytes + 255) & ~(size_t)255; return o; };
  __bf16* WOUT  = (__bf16*)(ws + alloc((size_t)VOCABN * HIDDENN * 2));   // 32.8 MB, L2-resident
  __bf16* BCAT  = (__bf16*)(ws + alloc((size_t)HIDDENN * KMID * 2));     // 2 MB
  __bf16* ABF   = (__bf16*)(ws + alloc((size_t)BATCHN * KMID * 2));      // 256 KB
  __bf16* HBF   = (__bf16*)(ws + alloc((size_t)BATCHN * HIDDENN * 2));   // 64 KB
  float*  HIDA  = (float*)(ws + alloc((size_t)BATCHN * HIDDENN * 4));
  float*  HIDB  = (float*)(ws + alloc((size_t)BATCHN * HIDDENN * 4));
  float*  STA   = (float*)(ws + alloc((size_t)BATCHN * NSTACKN * SSIZEN * ESIZEN * 4));
  float*  STB   = (float*)(ws + alloc((size_t)BATCHN * NSTACKN * SSIZEN * ESIZEN * 4));
  float*  ACTS  = (float*)(ws + alloc((size_t)BATCHN * NSTACKN * NACTN * 4));
  float*  PUSH  = (float*)(ws + alloc((size_t)BATCHN * NSTACKN * ESIZEN * 4));
  float*  BIASC = (float*)(ws + alloc((size_t)HIDDENN * 4));
  float*  LSE   = (float*)(ws + alloc((size_t)BATCHN * 4));

  float* hidbuf[2] = {HIDA, HIDB};
  float* stbuf[2]  = {STA, STB};
  float* outp = (float*)d_out;

  prep_kernel<<<2048, 256, 0, stream>>>(W_out, W_i2h, W_h2h, W_s2h, b_i2h, b_h2h,
                                        b_s2h, empty_elem, WOUT, BCAT, BIASC, STA);

  for (int t = 0; t < SEQN; ++t) {
    const float* hid_in = (t == 0) ? hidden0 : hidbuf[(t + 1) & 1];
    float* hid_out = hidbuf[t & 1];
    const float* st_in = stbuf[t & 1];
    float* st_out = stbuf[(t + 1) & 1];
    float* out_t = outp + (long)t * BATCHN * VOCABN;

    gates_kernel<<<BATCHN, 256, 0, stream>>>(hid_in, W_h2s, b_h2s, W_h2a, b_h2a, PUSH, ACTS);
    aprep_kernel<<<(BATCHN * KMID) / 256, 256, 0, stream>>>(inputs + t * BATCHN, embedding,
                                                            hid_in, st_in, ABF);
    stack_kernel<<<(BATCHN * NSTACKN * SSIZEN * ESIZEN) / 256, 256, 0, stream>>>(
        st_in, ACTS, PUSH, empty_elem, st_out);
    midgemm_kernel<<<4, 256, 0, stream>>>(ABF, BCAT, BIASC, hid_out, HBF);
    // 500 waves: 125 blocks x 128 threads (4 waves each)
    logits_kernel<<<125, 128, 0, stream>>>(HBF, WOUT, b_out, out_t);
    rowstats_kernel<<<BATCHN, 256, 0, stream>>>(out_t, LSE);
    logsub_kernel<<<2048, 256, 0, stream>>>(out_t, LSE);
  }
  copyhid_kernel<<<(BATCHN * HIDDENN) / 256, 256, 0, stream>>>(
      hidbuf[1], outp + (long)SEQN * BATCHN * VOCABN);
}
